// Transformer_72619307041471
// MI455X (gfx1250) — compile-verified, matrix operand order checked by reading, not run on hardware
//
#include <hip/hip_runtime.h>
#include <cmath>

// ---------------------------------------------------------------------------
// Types for CDNA5 WMMA (gfx1250, wave32)
// ---------------------------------------------------------------------------
typedef __bf16 bf16;
typedef __attribute__((ext_vector_type(8)))  __bf16 v8bf;
typedef __attribute__((ext_vector_type(16))) __bf16 v16bf;
typedef __attribute__((ext_vector_type(8)))  float  v8f;

// Problem constants (match the reference)
constexpr int B_  = 2;
constexpr int S_  = 1024;
constexpr int D_  = 1024;
constexpr int H_  = 16;
constexpr int HD_ = 64;
constexpr int L_  = 2;
constexpr int V_  = 32000;

// ---------------------------------------------------------------------------
// Generic NT GEMM:  C[M,N] (f32) = A[M,K](bf16) * Bm[N,K](bf16)^T
// Block tile 128x128, K-step 32, 256 threads (8 wave32), wave tile 32x64.
// Template: ACC_ADD -> C += acc (residual), GELU_ACT -> exact GELU epilogue.
// Optional Cb: mirrored bf16 output (same ldc layout).
// Batched over blockIdx.z with element strides sA/sB/sC; if bMod != 0 the
// B-operand batch offset is (z % bMod) * sB (shared rel_p across batch).
// ---------------------------------------------------------------------------
template <bool ACC_ADD, bool GELU_ACT>
__global__ __launch_bounds__(256) void gemm_bf16_nt(
    const bf16* __restrict__ A, const bf16* __restrict__ Bm,
    float* __restrict__ C, bf16* __restrict__ Cb,
    int M, int N, int K, int lda, int ldb, int ldc,
    long long sA, long long sB, long long sC, int bMod)
{
  __shared__ bf16 As[128][40];
  __shared__ bf16 Bs[128][40];

  const int z = blockIdx.z;
  const long long zb = bMod ? (long long)(z % bMod) : (long long)z;
  A  += (long long)z * sA;
  Bm += zb * sB;
  const long long cBase = (long long)z * sC;

  const int blockN = blockIdx.x * 128;
  const int blockM = blockIdx.y * 128;

  const int t    = threadIdx.x;
  const int lane = t & 31;
  const int wave = t >> 5;
  const int waveM = (wave & 3) * 32;   // 4 waves down M
  const int waveN = (wave >> 2) * 64;  // 2 waves across N
  const int l16 = lane & 15;
  const int hi8 = (lane >> 4) << 3;    // 0 or 8

  // global->LDS loader mapping: 2 threads per 32-elem row, 16 bf16 each
  const int ldRow = t >> 1;
  const int ldCol = (t & 1) << 4;

  v8f acc[2][4];
#pragma unroll
  for (int i = 0; i < 2; ++i)
#pragma unroll
    for (int j = 0; j < 4; ++j)
      acc[i][j] = (v8f){0.f, 0.f, 0.f, 0.f, 0.f, 0.f, 0.f, 0.f};

  const uint4 zero4 = make_uint4(0u, 0u, 0u, 0u);

  for (int k0 = 0; k0 < K; k0 += 32) {
    // ---- stage A tile ----
    {
      const int gr = blockM + ldRow;
      uint4 d0 = zero4, d1 = zero4;
      if (gr < M) {
        const uint4* gp = (const uint4*)(A + (long long)gr * lda + k0 + ldCol);
        d0 = gp[0];
        d1 = gp[1];
        if (k0 + 32 < K)  // next-tile prefetch -> global_prefetch_b8
          __builtin_prefetch((const void*)(A + (long long)gr * lda + k0 + 32 + ldCol), 0, 1);
      }
      uint4* sp = (uint4*)&As[ldRow][ldCol];
      sp[0] = d0;
      sp[1] = d1;
    }
    // ---- stage B tile ----
    {
      const int gr = blockN + ldRow;
      uint4 d0 = zero4, d1 = zero4;
      if (gr < N) {
        const uint4* gp = (const uint4*)(Bm + (long long)gr * ldb + k0 + ldCol);
        d0 = gp[0];
        d1 = gp[1];
        if (k0 + 32 < K)
          __builtin_prefetch((const void*)(Bm + (long long)gr * ldb + k0 + 32 + ldCol), 0, 1);
      }
      uint4* sp = (uint4*)&Bs[ldRow][ldCol];
      sp[0] = d0;
      sp[1] = d1;
    }
    __syncthreads();

    // ---- fragments (ISA 7.12.2: lane holds two 8-half contiguous K chunks) ----
    v16bf af[2];
#pragma unroll
    for (int tm = 0; tm < 2; ++tm) {
      const bf16* p = &As[waveM + tm * 16 + l16][hi8];
      v8bf lo = *(const v8bf*)p;
      v8bf hi = *(const v8bf*)(p + 16);
      af[tm] = __builtin_shufflevector(lo, hi, 0, 1, 2, 3, 4, 5, 6, 7,
                                               8, 9, 10, 11, 12, 13, 14, 15);
    }
    v16bf bfr[4];
#pragma unroll
    for (int tn = 0; tn < 4; ++tn) {
      const bf16* p = &Bs[waveN + tn * 16 + l16][hi8];
      v8bf lo = *(const v8bf*)p;
      v8bf hi = *(const v8bf*)(p + 16);
      bfr[tn] = __builtin_shufflevector(lo, hi, 0, 1, 2, 3, 4, 5, 6, 7,
                                                8, 9, 10, 11, 12, 13, 14, 15);
    }

#pragma unroll
    for (int tm = 0; tm < 2; ++tm)
#pragma unroll
      for (int tn = 0; tn < 4; ++tn)
        acc[tm][tn] = __builtin_amdgcn_wmma_f32_16x16x32_bf16(
            false, af[tm], false, bfr[tn], (short)0, acc[tm][tn], false, false);

    __syncthreads();
  }

  // ---- epilogue (C layout per ISA: lane%16 = N, vgpr r + 8*(lane/16) = M) ----
  const int mBase = blockM + waveM + ((lane >> 4) << 3);
  const int nBase = blockN + waveN + l16;
#pragma unroll
  for (int tm = 0; tm < 2; ++tm) {
#pragma unroll
    for (int tn = 0; tn < 4; ++tn) {
      const int col = nBase + tn * 16;
      if (col >= N) continue;
#pragma unroll
      for (int r = 0; r < 8; ++r) {
        const int row = mBase + tm * 16 + r;
        if (row >= M) continue;
        float v = acc[tm][tn][r];
        if (GELU_ACT) v = 0.5f * v * (1.0f + erff(v * 0.70710678118654752f));
        const long long ci = cBase + (long long)row * ldc + col;
        if (ACC_ADD) v += C[ci];
        C[ci] = v;
        if (Cb) Cb[ci] = (bf16)v;
      }
    }
  }
}

// ---------------------------------------------------------------------------
// Elementwise / glue kernels
// ---------------------------------------------------------------------------
__global__ void embed_kernel(const int* __restrict__ tokens,
                             const float* __restrict__ embed,
                             float* __restrict__ x)
{
  long long i = (long long)blockIdx.x * blockDim.x + threadIdx.x;
  if (i >= (long long)B_ * S_ * D_) return;
  const int row = (int)(i >> 10);
  const int d = (int)(i & (D_ - 1));
  x[i] = embed[(long long)tokens[row] * D_ + d];
}

__global__ void f32_to_bf16_kernel(const float* __restrict__ src,
                                   bf16* __restrict__ dst, long long n)
{
  long long i = (long long)blockIdx.x * blockDim.x + threadIdx.x;
  if (i < n) dst[i] = (bf16)src[i];
}

// rel = sinusoid_matrix(S, D)[::-1], stored bf16 directly (GEMM A operand)
__global__ void rel_kernel(bf16* __restrict__ rel)
{
  long long i = (long long)blockIdx.x * blockDim.x + threadIdx.x;
  if (i >= (long long)S_ * D_) return;
  const int j = (int)(i & (D_ - 1));
  const int r = (int)(i >> 10);
  const int p = S_ - 1 - r;
  const int tt = j >> 1;
  const float w = powf(10000.0f, -2.0f * (float)tt / (float)D_);
  const float c = (float)p * w;
  rel[i] = (bf16)((j & 1) ? cosf(c) : sinf(c));
}

// RMSNorm (p=1.0): one 256-thread block per row, writes bf16 for the GEMM
__global__ __launch_bounds__(256) void rmsnorm_kernel(
    const float* __restrict__ x, const float* __restrict__ scale,
    bf16* __restrict__ out)
{
  __shared__ float red[256];
  const int row = blockIdx.x;
  const float* xr = x + (long long)row * D_;
  float ss = 0.f;
#pragma unroll
  for (int i = 0; i < D_ / 256; ++i) {
    const float v = xr[threadIdx.x + i * 256];
    ss += v * v;
  }
  red[threadIdx.x] = ss;
  __syncthreads();
  for (int s = 128; s > 0; s >>= 1) {
    if (threadIdx.x < s) red[threadIdx.x] += red[threadIdx.x + s];
    __syncthreads();
  }
  const float rms = sqrtf(red[0]) * (1.0f / 32.0f);  // /sqrt(D)
  const float inv = 1.0f / (rms + 1e-6f);
  bf16* orow = out + (long long)row * D_;
#pragma unroll
  for (int i = 0; i < D_ / 256; ++i) {
    const int j = threadIdx.x + i * 256;
    orow[j] = (bf16)(xr[j] * inv * scale[j]);
  }
}

// Split heads, add pu/pv to q, convert to bf16; v stored transposed [B,H,HD,S]
__global__ void split_heads_kernel(
    const float* __restrict__ qkv, const float* __restrict__ pu,
    const float* __restrict__ pv, bf16* __restrict__ qpu,
    bf16* __restrict__ qpv, bf16* __restrict__ kbf, bf16* __restrict__ vT)
{
  long long idx = (long long)blockIdx.x * blockDim.x + threadIdx.x;
  if (idx >= (long long)B_ * H_ * S_ * HD_) return;
  const int hd = (int)(idx & (HD_ - 1));
  const int s  = (int)((idx >> 6) & (S_ - 1));
  const int h  = (int)((idx >> 16) & (H_ - 1));
  const int b  = (int)(idx >> 20);
  const long long base = (long long)(b * S_ + s) * (3 * D_) + h * HD_ + hd;
  const float q = qkv[base];
  const float k = qkv[base + D_];
  const float v = qkv[base + 2 * D_];
  qpu[idx] = (bf16)(q + pu[h * HD_ + hd]);
  qpv[idx] = (bf16)(q + pv[h * HD_ + hd]);
  kbf[idx] = (bf16)k;
  vT[((long long)(b * H_ + h) * HD_ + hd) * S_ + s] = (bf16)v;
}

// Softmax with TXL rel-shift fold-in, causal mask, 1/sqrt(HD) scale -> bf16
__global__ __launch_bounds__(256) void softmax_kernel(
    const float* __restrict__ dots, const float* __restrict__ qp,
    bf16* __restrict__ probs)
{
  __shared__ float red[256];
  const long long rid = blockIdx.x;  // over B*H*S rows
  const int i = (int)(rid & (S_ - 1));
  const float* drow = dots + rid * S_;
  const float* qrow = qp + (rid >> 10) * (long long)S_ * S_ + (long long)i * S_;
  bf16* prow = probs + rid * S_;
  const int t = threadIdx.x;

  float vals[4];
  float mx = -INFINITY;
#pragma unroll
  for (int c = 0; c < 4; ++c) {
    const int j = t + c * 256;
    float v;
    if (j > i) {
      v = -INFINITY;
    } else {
      const int js = (j - i - 1 + S_) & (S_ - 1);  // (j-i-1) mod S
      v = (drow[j] + qrow[js]) * 0.125f;           // 1/sqrt(64)
    }
    vals[c] = v;
    mx = fmaxf(mx, v);
  }
  red[t] = mx;
  __syncthreads();
  for (int s = 128; s > 0; s >>= 1) {
    if (t < s) red[t] = fmaxf(red[t], red[t + s]);
    __syncthreads();
  }
  mx = red[0];
  __syncthreads();
  float sum = 0.f;
#pragma unroll
  for (int c = 0; c < 4; ++c) {
    const float e = expf(vals[c] - mx);
    vals[c] = e;
    sum += e;
  }
  red[t] = sum;
  __syncthreads();
  for (int s = 128; s > 0; s >>= 1) {
    if (t < s) red[t] += red[t + s];
    __syncthreads();
  }
  const float inv = 1.0f / red[0];
#pragma unroll
  for (int c = 0; c < 4; ++c) {
    const int j = t + c * 256;
    prow[j] = (bf16)(vals[c] * inv);
  }
}

// [B,H,S,HD] f32 -> [B,S,D] bf16 (merge heads for out-proj GEMM)
__global__ void merge_heads_kernel(const float* __restrict__ attnO,
                                   bf16* __restrict__ out)
{
  long long idx = (long long)blockIdx.x * blockDim.x + threadIdx.x;
  if (idx >= (long long)B_ * S_ * D_) return;
  const int d = (int)(idx & (D_ - 1));
  const int s = (int)((idx >> 10) & (S_ - 1));
  const int b = (int)(idx >> 20);
  const int h = d >> 6;
  const int hd = d & (HD_ - 1);
  out[idx] = (bf16)attnO[((long long)(b * H_ + h) * S_ + s) * HD_ + hd];
}

// ---------------------------------------------------------------------------
// Orchestration
// ---------------------------------------------------------------------------
extern "C" void kernel_launch(void* const* d_in, const int* in_sizes, int n_in,
                              void* d_out, int out_size, void* d_ws,
                              size_t ws_size, hipStream_t stream)
{
  (void)in_sizes; (void)n_in; (void)out_size; (void)ws_size;

  const int*   tokens  = (const int*)d_in[0];
  const float* embed   = (const float*)d_in[1];
  const float* qkv_w   = (const float*)d_in[2];
  const float* out_w   = (const float*)d_in[3];
  const float* pk_w    = (const float*)d_in[4];
  const float* pu      = (const float*)d_in[5];
  const float* pv      = (const float*)d_in[6];
  const float* ln1_s   = (const float*)d_in[7];
  const float* ln2_s   = (const float*)d_in[8];
  const float* w1      = (const float*)d_in[9];
  const float* w2      = (const float*)d_in[10];
  const float* unembed = (const float*)d_in[11];
  float* out = (float*)d_out;

  char* ws = (char*)d_ws;
  auto MB = [](size_t m) { return m << 20; };
  float* x     = (float*)(ws + MB(0));    // [B,S,D] f32          8 MB
  bf16*  hb    = (bf16*)(ws + MB(8));     // normed act bf16      4 MB
  float* qkv   = (float*)(ws + MB(12));   // [B,S,3D] f32        24 MB
  bf16*  wbuf  = (bf16*)(ws + MB(36));    // converted weights    8 MB
  bf16*  qpu   = (bf16*)(ws + MB(44));    // [B,H,S,HD]           4 MB
  bf16*  qpv   = (bf16*)(ws + MB(48));    //                      4 MB
  bf16*  kbf   = (bf16*)(ws + MB(52));    //                      4 MB
  bf16*  vT    = (bf16*)(ws + MB(56));    // [B,H,HD,S]           4 MB
  bf16*  relb  = (bf16*)(ws + MB(60));    // [S,D] bf16           2 MB
  bf16*  relpb = (bf16*)(ws + MB(62));    // rel_p bf16           2 MB
  float* relpf = (float*)(ws + MB(64));   // rel_p f32            4 MB
  float* attnO = (float*)(ws + MB(68));   // [B,H,S,HD] f32       8 MB
  bf16*  attnM = (bf16*)(ws + MB(76));    // [B,S,D] bf16         4 MB
  bf16*  xb    = (bf16*)(ws + MB(80));    // final x bf16         4 MB
  bf16*  unb   = (bf16*)(ws + MB(84));    // unembed bf16        64 MB
  float* dots  = (float*)(ws + MB(148));  // [B,H,S,S] f32      128 MB
  float* qp    = (float*)(ws + MB(276));  // [B,H,S,S] f32      128 MB
  bf16*  probs = (bf16*)(ws + MB(404));   // [B,H,S,S] bf16      64 MB
  float* midf  = (float*)(ws + MB(468));  // [B,S,4D] f32        32 MB
  bf16*  midb  = (bf16*)(ws + MB(500));   // [B,S,4D] bf16       16 MB

  const long long nBSD = (long long)B_ * S_ * D_;
  const int TPB = 256;
  auto blks = [&](long long n) { return (unsigned)((n + TPB - 1) / TPB); };

  // One-time prep
  rel_kernel<<<blks((long long)S_ * D_), TPB, 0, stream>>>(relb);
  embed_kernel<<<blks(nBSD), TPB, 0, stream>>>(tokens, embed, x);
  f32_to_bf16_kernel<<<blks((long long)V_ * D_), TPB, 0, stream>>>(
      unembed, unb, (long long)V_ * D_);

  for (int l = 0; l < L_; ++l) {
    const float* qkvw_l = qkv_w + (long long)l * 3 * D_ * D_;
    const float* outw_l = out_w + (long long)l * D_ * D_;
    const float* pkw_l  = pk_w + (long long)l * D_ * D_;
    const float* pu_l   = pu + (long long)l * H_ * HD_;
    const float* pv_l   = pv + (long long)l * H_ * HD_;
    const float* w1_l   = w1 + (long long)l * 4 * D_ * D_;
    const float* w2_l   = w2 + (long long)l * D_ * 4 * D_;

    // ---- attention block ----
    rmsnorm_kernel<<<B_ * S_, TPB, 0, stream>>>(x, ln1_s + l * D_, hb);

    f32_to_bf16_kernel<<<blks((long long)3 * D_ * D_), TPB, 0, stream>>>(
        qkvw_l, wbuf, (long long)3 * D_ * D_);
    {
      dim3 g((3 * D_) / 128, (B_ * S_) / 128, 1);
      gemm_bf16_nt<false, false><<<g, TPB, 0, stream>>>(
          hb, wbuf, qkv, nullptr, B_ * S_, 3 * D_, D_, D_, D_, 3 * D_,
          0LL, 0LL, 0LL, 0);
    }
    split_heads_kernel<<<blks(nBSD), TPB, 0, stream>>>(qkv, pu_l, pv_l, qpu,
                                                       qpv, kbf, vT);
    {  // dots[b,h] = (q+pu) @ k^T   (batched over B*H)
      dim3 g(S_ / 128, S_ / 128, B_ * H_);
      gemm_bf16_nt<false, false><<<g, TPB, 0, stream>>>(
          qpu, kbf, dots, nullptr, S_, S_, HD_, HD_, HD_, S_,
          (long long)S_ * HD_, (long long)S_ * HD_, (long long)S_ * S_, 0);
    }
    f32_to_bf16_kernel<<<blks((long long)D_ * D_), TPB, 0, stream>>>(
        pkw_l, wbuf, (long long)D_ * D_);
    {  // rel_p = rel @ pk_w^T (also mirrored to bf16)
      dim3 g(D_ / 128, S_ / 128, 1);
      gemm_bf16_nt<false, false><<<g, TPB, 0, stream>>>(
          relb, wbuf, relpf, relpb, S_, D_, D_, D_, D_, D_, 0LL, 0LL, 0LL, 0);
    }
    {  // qp[b,h] = (q+pv) @ rel_p[:, h*HD:(h+1)*HD]^T  (B offset = (z%H)*HD)
      dim3 g(S_ / 128, S_ / 128, B_ * H_);
      gemm_bf16_nt<false, false><<<g, TPB, 0, stream>>>(
          qpv, relpb, qp, nullptr, S_, S_, HD_, HD_, D_, S_,
          (long long)S_ * HD_, (long long)HD_, (long long)S_ * S_, H_);
    }
    softmax_kernel<<<B_ * H_ * S_, TPB, 0, stream>>>(dots, qp, probs);
    {  // attn[b,h] = probs @ v   (v pre-transposed -> NT)
      dim3 g((HD_ + 127) / 128, S_ / 128, B_ * H_);
      gemm_bf16_nt<false, false><<<g, TPB, 0, stream>>>(
          probs, vT, attnO, nullptr, S_, HD_, S_, S_, S_, HD_,
          (long long)S_ * S_, (long long)HD_ * S_, (long long)S_ * HD_, 0);
    }
    merge_heads_kernel<<<blks(nBSD), TPB, 0, stream>>>(attnO, attnM);
    f32_to_bf16_kernel<<<blks((long long)D_ * D_), TPB, 0, stream>>>(
        outw_l, wbuf, (long long)D_ * D_);
    {  // x += attn @ out_w^T  (residual in epilogue)
      dim3 g(D_ / 128, (B_ * S_) / 128, 1);
      gemm_bf16_nt<true, false><<<g, TPB, 0, stream>>>(
          attnM, wbuf, x, nullptr, B_ * S_, D_, D_, D_, D_, D_,
          0LL, 0LL, 0LL, 0);
    }

    // ---- FFN block ----
    rmsnorm_kernel<<<B_ * S_, TPB, 0, stream>>>(x, ln2_s + l * D_, hb);
    f32_to_bf16_kernel<<<blks((long long)4 * D_ * D_), TPB, 0, stream>>>(
        w1_l, wbuf, (long long)4 * D_ * D_);
    {  // mid = gelu(h2 @ w1^T), mirrored to bf16 for w2 GEMM
      dim3 g((4 * D_) / 128, (B_ * S_) / 128, 1);
      gemm_bf16_nt<false, true><<<g, TPB, 0, stream>>>(
          hb, wbuf, midf, midb, B_ * S_, 4 * D_, D_, D_, D_, 4 * D_,
          0LL, 0LL, 0LL, 0);
    }
    f32_to_bf16_kernel<<<blks((long long)4 * D_ * D_), TPB, 0, stream>>>(
        w2_l, wbuf, (long long)D_ * 4 * D_);
    {  // x += mid @ w2^T
      dim3 g(D_ / 128, (B_ * S_) / 128, 1);
      gemm_bf16_nt<true, false><<<g, TPB, 0, stream>>>(
          midb, wbuf, x, nullptr, B_ * S_, D_, 4 * D_, 4 * D_, 4 * D_, D_,
          0LL, 0LL, 0LL, 0);
    }
  }

  // ---- logits = x @ unembed^T (dominant GEMM: 2048 x 32000 x 1024) ----
  f32_to_bf16_kernel<<<blks(nBSD), TPB, 0, stream>>>(x, xb, nBSD);
  {
    dim3 g((V_ + 127) / 128, (B_ * S_) / 128, 1);
    gemm_bf16_nt<false, false><<<g, TPB, 0, stream>>>(
        xb, unb, out, nullptr, B_ * S_, V_, D_, D_, D_, V_, 0LL, 0LL, 0LL, 0);
  }
}
